// AttackModule_75539884802062
// MI455X (gfx1250) — matrix operand order repaired and checked by reading, not compiled, not run on hardware
//
#include <hip/hip_runtime.h>
#include <hip/hip_bf16.h>
#include <math.h>

// Problem constants (from reference)
#define B_     32
#define S_     64
#define E_     512
#define V_     50257
#define M_     2048            // B*S token rows
#define KTOP_  32              // int(0.5 * S)
#define EPS_   0.4f
#define STDM_  3.0f

// GEMM tiling: block = 8 waves; each wave: one 16-row m-tile x 128 cols (8 v-tiles).
// All 8 waves share a 128-column B strip staged in LDS (double buffered).
#define VSTRIPS_   393                 // ceil(ceil(V/16)/8) strips of 128 columns
#define VPAD_      (VSTRIPS_ * 128)    // 50304 padded vocab rows
#define MBLK_      128                 // rows per block (8 waves x 16)
#define LDSSTRIDE_ 80                  // bytes per 32-K column chunk (64 data + 16 pad)
#define LDSBUF_    (128 * LDSSTRIDE_)  // 10240 bytes per buffer

typedef __attribute__((ext_vector_type(16))) __bf16 v16bf;
typedef __attribute__((ext_vector_type(8)))  __bf16 v8bf;
typedef __attribute__((ext_vector_type(8)))  float  v8f;

union V16U { v16bf v; v8bf h[2]; };

// CDNA5 async global->LDS copies (ASYNCcnt-tracked), with safe fallback.
// Probe-verified: the builtin takes GCC-style int4* (generic) pointers.
#if __has_builtin(__builtin_amdgcn_global_load_async_to_lds_b128) && \
    __has_builtin(__builtin_amdgcn_s_wait_asynccnt)
#define USE_ASYNC 1
typedef int v4i_ __attribute__((__vector_size__(16)));
#else
#define USE_ASYNC 0
#endif

// ---------------- reduction helper (256-thread blocks) ----------------
__device__ __forceinline__ float blockSum256(float v) {
    __shared__ float sm[256];
    __syncthreads();
    sm[threadIdx.x] = v;
    __syncthreads();
    for (int s = 128; s > 0; s >>= 1) {
        if ((int)threadIdx.x < s) sm[threadIdx.x] += sm[threadIdx.x + s];
        __syncthreads();
    }
    return sm[0];
}

// ---------------- phase 1: abs-grad per token ----------------
__global__ void k_absgrad(const float* __restrict__ g, float* __restrict__ absg) {
    int m = blockIdx.x;
    const float* row = g + (size_t)m * E_;
    float a = fabsf(row[threadIdx.x]) + fabsf(row[threadIdx.x + 256]);
    float s = blockSum256(a);
    if (threadIdx.x == 0) absg[m] = s;
}

// ---------------- phase 2: per-batch stable top-k mask ----------------
__global__ void k_topk(const float* __restrict__ absg, int* __restrict__ mask) {
    __shared__ float vals[S_];
    int b = blockIdx.x, t = threadIdx.x;
    vals[t] = absg[b * S_ + t];
    __syncthreads();
    float v = vals[t];
    int rank = 0;
    for (int j = 0; j < S_; ++j) {
        float vj = vals[j];
        rank += (vj > v) || (vj == v && j < t);   // stable top_k semantics
    }
    mask[b * S_ + t] = (rank < KTOP_) ? 1 : 0;
}

// ---------------- phase 3: masked sum / sumsq partials ----------------
__global__ void k_stats_part(const float* __restrict__ g, const int* __restrict__ mask,
                             float* __restrict__ s1, float* __restrict__ s2) {
    int m = blockIdx.x;
    float a = 0.f, b = 0.f;
    if (mask[m]) {
        const float* row = g + (size_t)m * E_;
        float x0 = row[threadIdx.x], x1 = row[threadIdx.x + 256];
        a = x0 + x1;
        b = x0 * x0 + x1 * x1;
    }
    float sa = blockSum256(a);
    float sb = blockSum256(b);
    if (threadIdx.x == 0) { s1[m] = sa; s2[m] = sb; }
}

__global__ void k_stats_final(const float* __restrict__ s1, const float* __restrict__ s2,
                              float* __restrict__ stats) {
    float a = 0.f, b = 0.f;
    for (int i = threadIdx.x; i < M_; i += 256) { a += s1[i]; b += s2[i]; }
    float S1 = blockSum256(a);
    float S2 = blockSum256(b);
    if (threadIdx.x == 0) {
        const float n = 32.0f * 32.0f * 512.0f;               // B*k*E = 524288
        float mean = S1 / n;
        float var  = (S2 - S1 * S1 / n) / (n - 1.0f);         // unbiased
        float sd   = sqrtf(var);
        stats[0] = mean - STDM_ * sd;                         // lb
        stats[1] = mean + STDM_ * sd;                         // ub
    }
}

// ---------------- phase 4: perturbed tokens (bf16) + 1/p_norm ----------------
__global__ void k_perturb(const int* __restrict__ utt, const float* __restrict__ emb,
                          const float* __restrict__ g, const int* __restrict__ mask,
                          const float* __restrict__ stats,
                          __bf16* __restrict__ A, float* __restrict__ p_inv) {
    int m   = blockIdx.x;
    int tok = utt[m];
    float lb = stats[0], ub = stats[1];
    int msk = mask[m];
    const float* erow = emb + (size_t)tok * E_;
    const float* grow = g   + (size_t)m   * E_;
    float nrm = 0.f;
#pragma unroll
    for (int i = 0; i < 2; ++i) {
        int e = threadIdx.x + i * 256;
        float p  = erow[e];
        float gr = grow[e];
        if (msk && (gr < lb || gr > ub))
            p += EPS_ * ((gr > 0.f) ? 1.f : ((gr < 0.f) ? -1.f : 0.f));
        A[(size_t)m * E_ + e] = (__bf16)p;
        nrm += p * p;                                         // norm in f32 (pre-bf16)
    }
    float s = blockSum256(nrm);
    if (threadIdx.x == 0) p_inv[m] = 1.0f / fmaxf(sqrtf(s), 1e-8f);
}

// ---------------- phase 5: emb table -> bf16 (zero-padded) + 1/t_norm ----------------
__global__ void k_embprep(const float* __restrict__ emb, __bf16* __restrict__ Bm,
                          float* __restrict__ t_inv) {
    int v = blockIdx.x;
    if (v < V_) {
        const float* erow = emb + (size_t)v * E_;
        float nrm = 0.f;
#pragma unroll
        for (int i = 0; i < 2; ++i) {
            int e = threadIdx.x + i * 256;
            float x = erow[e];
            Bm[(size_t)v * E_ + e] = (__bf16)x;
            nrm += x * x;
        }
        float s = blockSum256(nrm);
        if (threadIdx.x == 0) t_inv[v] = 1.0f / fmaxf(sqrtf(s), 1e-8f);
    } else {
#pragma unroll
        for (int i = 0; i < 2; ++i)
            Bm[(size_t)v * E_ + threadIdx.x + i * 256] = (__bf16)0.0f;
        if (threadIdx.x == 0) t_inv[v] = 0.f;
    }
}

__global__ void k_zero_amax(unsigned long long* __restrict__ amax) {
    int i = blockIdx.x * blockDim.x + threadIdx.x;
    if (i < M_) amax[i] = 0ull;
}

// ---------------- phase 6: LDS-staged bf16 WMMA GEMM + scale + fused argmax ----------------
__device__ __forceinline__ unsigned ordf(float f) {
    unsigned x = __float_as_uint(f);
    return (x & 0x80000000u) ? ~x : (x | 0x80000000u);        // monotonic float->u32
}

__global__ void __launch_bounds__(256)
k_gemm(const __bf16* __restrict__ A, const __bf16* __restrict__ Bm,
       const float* __restrict__ p_inv, const float* __restrict__ t_inv,
       float* __restrict__ sim, unsigned long long* __restrict__ amax) {
    __shared__ __align__(16) char lds[2 * LDSBUF_];
    const int tid  = threadIdx.x;
    const int wave = tid >> 5;
    const int lane = tid & 31;
    const int h    = lane >> 4;                 // lane half
    const int n    = lane & 15;
    const int m0   = blockIdx.y * MBLK_ + wave * 16;   // per-wave m-tile
    const int v0   = blockIdx.x * 128;                 // per-block v strip

    // Staging role: thread copies 32 bytes of one column's 32-K chunk per step.
    const int    sc    = tid >> 1;              // column 0..127
    const int    spart = (tid & 1);             // which 16-element half of the chunk
    const __bf16* bcol = Bm + (size_t)(v0 + sc) * E_ + spart * 16;
    const int    soff  = sc * LDSSTRIDE_ + spart * 32;

    // Stage k-step 0 into buffer 0
    {
        char* d = lds + soff;
#if USE_ASYNC
        __builtin_amdgcn_global_load_async_to_lds_b128((v4i_*)bcol, (v4i_*)d, 0, 0);
        __builtin_amdgcn_global_load_async_to_lds_b128((v4i_*)(bcol + 8), (v4i_*)(d + 16), 0, 0);
#else
        uint4 d0 = *(const uint4*)bcol, d1 = *(const uint4*)(bcol + 8);
        *(uint4*)d = d0; *(uint4*)(d + 16) = d1;
#endif
    }

    v8f acc[8] = {{}, {}, {}, {}, {}, {}, {}, {}};
    const __bf16* arow = A + (size_t)(m0 + n) * E_;    // A: M = lane&15

    for (int ks = 0; ks < 16; ++ks) {
        const int kk  = ks * 32;
        const int cur = ks & 1;
#if USE_ASYNC
        __builtin_amdgcn_s_wait_asynccnt(0);
#endif
        __syncthreads();
        if (ks < 15) {                                 // stage next chunk (other buffer)
            const __bf16* s = bcol + kk + 32;
            char* d = lds + (cur ^ 1) * LDSBUF_ + soff;
#if USE_ASYNC
            __builtin_amdgcn_global_load_async_to_lds_b128((v4i_*)s, (v4i_*)d, 0, 0);
            __builtin_amdgcn_global_load_async_to_lds_b128((v4i_*)(s + 8), (v4i_*)(d + 16), 0, 0);
#else
            uint4 d0 = *(const uint4*)s, d1 = *(const uint4*)(s + 8);
            *(uint4*)d = d0; *(uint4*)(d + 16) = d1;
#endif
        }
        V16U a;                                        // 16x32 bf16 A operand
        a.h[0] = *(const v8bf*)(arow + kk + h * 16);
        a.h[1] = *(const v8bf*)(arow + kk + h * 16 + 8);
        const char* bb = lds + cur * LDSBUF_;

        // Load ALL 8 B operands into distinct registers first, then issue the
        // 8 WMMAs back-to-back: lets the scheduler batch ds_loads and avoids a
        // ds_load -> s_wait_dscnt -> wmma serialization per tile.
        V16U bop[8];
#pragma unroll
        for (int t = 0; t < 8; ++t) {
            const char* p = bb + (t * 16 + n) * LDSSTRIDE_ + h * 32;
            bop[t].h[0] = *(const v8bf*)p;
            bop[t].h[1] = *(const v8bf*)(p + 16);
        }
#pragma unroll
        for (int t = 0; t < 8; ++t) {
            acc[t] = __builtin_amdgcn_wmma_f32_16x16x32_bf16(
                         false, a.v, false, bop[t].v, (short)0, acc[t], false, false);
        }
    }

    // Epilogue: C layout — elem r: M = r + 8*h, N = lane&15
    float pr[8];
#pragma unroll
    for (int r = 0; r < 8; ++r) pr[r] = p_inv[m0 + r + 8 * h];

    float    bestv[8];
    unsigned besti[8];
#pragma unroll
    for (int r = 0; r < 8; ++r) { bestv[r] = -__builtin_inff(); besti[r] = 0u; }

#pragma unroll
    for (int t = 0; t < 8; ++t) {
        int v = v0 + t * 16 + n;
        float tv = t_inv[v];
        bool valid = v < V_;
#pragma unroll
        for (int r = 0; r < 8; ++r) {
            float sv = acc[t][r] * pr[r] * tv;
            if (valid) sim[(size_t)(m0 + r + 8 * h) * V_ + v] = sv;
            float cand = valid ? sv : -__builtin_inff();
            if (cand > bestv[r]) { bestv[r] = cand; besti[r] = (unsigned)v; }
        }
    }

    // Per-row argmax: butterfly max over the 16-lane half, then one atomic
#pragma unroll
    for (int r = 0; r < 8; ++r) {
        unsigned long long key = ((unsigned long long)ordf(bestv[r]) << 32) |
                                 (unsigned long long)(0xFFFFFFFFu - besti[r]);
#pragma unroll
        for (int off = 1; off < 16; off <<= 1) {
            unsigned lo = (unsigned)key, hi = (unsigned)(key >> 32);
            lo = __shfl_xor(lo, off, 32);
            hi = __shfl_xor(hi, off, 32);
            unsigned long long other = ((unsigned long long)hi << 32) | lo;
            if (other > key) key = other;
        }
        if (n == 0) atomicMax(&amax[m0 + r + 8 * h], key);
    }
}

// ---------------- phase 7: unpack nearest-neighbor ids ----------------
__global__ void k_nn(const unsigned long long* __restrict__ amax, float* __restrict__ out_nn) {
    int i = blockIdx.x * blockDim.x + threadIdx.x;
    if (i < M_) {
        unsigned v = 0xFFFFFFFFu - (unsigned)(amax[i] & 0xFFFFFFFFull);
        out_nn[i] = (float)v;
    }
}

// ---------------- host launcher ----------------
extern "C" void kernel_launch(void* const* d_in, const int* in_sizes, int n_in,
                              void* d_out, int out_size, void* d_ws, size_t ws_size,
                              hipStream_t stream) {
    (void)in_sizes; (void)n_in; (void)out_size; (void)ws_size;
    const int*   utt = (const int*)d_in[0];    // [B*S]
    const float* emb = (const float*)d_in[1];  // [V*E]
    const float* g   = (const float*)d_in[2];  // [B*S*E]

    float* out_nn  = (float*)d_out;            // [M]  nn indices (as float)
    float* out_sim = (float*)d_out + M_;       // [M*V] similarities

    // Workspace carve-up (~54 MB total)
    char*  ws  = (char*)d_ws;
    size_t off = 0;
    auto alloc = [&](size_t bytes) -> void* {
        void* p = ws + off;
        off += (bytes + 255) & ~(size_t)255;
        return p;
    };
    __bf16* A     = (__bf16*)alloc((size_t)M_ * E_ * 2);        //  2.0 MB
    __bf16* Bm    = (__bf16*)alloc((size_t)VPAD_ * E_ * 2);     // 51.5 MB
    float*  p_inv = (float*) alloc((size_t)M_ * 4);
    float*  t_inv = (float*) alloc((size_t)VPAD_ * 4);
    float*  absg  = (float*) alloc((size_t)M_ * 4);
    int*    mask  = (int*)   alloc((size_t)M_ * 4);
    float*  s1    = (float*) alloc((size_t)M_ * 4);
    float*  s2    = (float*) alloc((size_t)M_ * 4);
    float*  stats = (float*) alloc(16);
    unsigned long long* amax = (unsigned long long*)alloc((size_t)M_ * 8);

    k_absgrad    <<<M_,    256, 0, stream>>>(g, absg);
    k_topk       <<<B_,    S_,  0, stream>>>(absg, mask);
    k_stats_part <<<M_,    256, 0, stream>>>(g, mask, s1, s2);
    k_stats_final<<<1,     256, 0, stream>>>(s1, s2, stats);
    k_perturb    <<<M_,    256, 0, stream>>>(utt, emb, g, mask, stats, A, p_inv);
    k_embprep    <<<VPAD_, 256, 0, stream>>>(emb, Bm, t_inv);
    k_zero_amax  <<<(M_ + 255) / 256, 256, 0, stream>>>(amax);
    k_gemm       <<<dim3(VSTRIPS_, 16), 256, 0, stream>>>(A, Bm, p_inv, t_inv, out_sim, amax);
    k_nn         <<<(M_ + 255) / 256, 256, 0, stream>>>(amax, out_nn);
}
